// SelfAttention_50044958933547
// MI455X (gfx1250) — matrix operand order repaired and checked by reading, not compiled, not run on hardware
//
#include <hip/hip_runtime.h>
#include <hip/hip_bf16.h>

// ---------------------------------------------------------------------------
// Self-attention forward on MI455X (gfx1250), bf16 WMMA pipeline:
//   0) one-shot fp32->bf16 conversion of inp; convert+transpose of all weights
//   1) Q/K/V = inp @ W{q,k,v}        (bf16 wmma GEMM, async-LDS staged tiles)
//   2) flash attention per (b,h,64q) (wmma QK^T + online softmax + wmma PV,
//                                     K & V tiles async-DMA'd to LDS)
//   3) out = O @ Wu + bu             (bf16 wmma GEMM, fp32 out)
// ---------------------------------------------------------------------------

#define B_   2
#define T_   2048
#define DIN  1024
#define EH   1024
#define E_   64
#define H_   16

typedef __attribute__((ext_vector_type(16))) __bf16 v16bf;
typedef __attribute__((ext_vector_type(8)))  __bf16 v8bf;
typedef __attribute__((ext_vector_type(8)))  float  v8f;
typedef __attribute__((ext_vector_type(4)))  int    v4i;

#if defined(__has_builtin)
#if __has_builtin(__builtin_amdgcn_global_load_async_to_lds_b128) && \
    __has_builtin(__builtin_amdgcn_s_wait_asynccnt)
#define HAVE_ASYNC_LDS 1
#endif
#endif

#ifdef HAVE_ASYNC_LDS
// Builtin signature (from hipcc diagnostic): param0 = v4i __device__(AS1)*,
// param1 = v4i __shared__(AS3)*, then imm offset, imm cpol.
#define GPTR(p) ((__attribute__((address_space(1))) v4i*)(p))
#define LPTR(p) ((__attribute__((address_space(3))) v4i*)(p))
#endif

// A-matrix 16x32 bf16 fragment: lane l -> m=l%16, hi=l/16,
// elems i = A[m][hi*16+i] (contiguous k order, matching make_bfrag).
static __device__ __forceinline__ v16bf make_afrag(const __bf16* row, int hi) {
  v8bf lo = *(const v8bf*)(row + hi * 16);
  v8bf up = *(const v8bf*)(row + hi * 16 + 8);
  v16bf a;
#pragma unroll
  for (int i = 0; i < 8; ++i) { a[i] = lo[i]; a[8 + i] = up[i]; }
  return a;
}

// B-matrix 32x16 bf16 fragment: lane l -> n=l%16, hi=l/16, elems i = B[hi*16+i][n].
// Caller passes a pointer to B^T[n] + hi*16 (16 contiguous bf16).
static __device__ __forceinline__ v16bf make_bfrag(const __bf16* colptr) {
  v8bf lo = *(const v8bf*)(colptr);
  v8bf up = *(const v8bf*)(colptr + 8);
  v16bf b;
#pragma unroll
  for (int i = 0; i < 8; ++i) { b[i] = lo[i]; b[8 + i] = up[i]; }
  return b;
}

static __device__ __forceinline__ v8f wmma_bf16(v16bf a, v16bf b, v8f c) {
  return __builtin_amdgcn_wmma_f32_16x16x32_bf16(false, a, false, b, (short)0, c,
                                                 false, false);
}

// ---------------------------------------------------------------------------
// One-shot fp32 -> bf16 elementwise conversion (n multiple of 2048)
// ---------------------------------------------------------------------------
__global__ __launch_bounds__(256) void convert_bf16(const float* __restrict__ src,
                                                    __bf16* __restrict__ dst) {
  int i = (blockIdx.x * 256 + threadIdx.x) * 8;
#pragma unroll
  for (int j = 0; j < 8; ++j) dst[i + j] = (__bf16)src[i + j];
}

// ---------------------------------------------------------------------------
// One-shot fp32 [K x N] -> bf16 W^T [N x K] convert + transpose (32x32 tiles)
// ---------------------------------------------------------------------------
__global__ __launch_bounds__(256) void transpose_convert(const float* __restrict__ W,
                                                         __bf16* __restrict__ Wt,
                                                         int K, int N) {
  __shared__ __bf16 tile[32][33];
  const int k0 = blockIdx.x * 32, n0 = blockIdx.y * 32;
#pragma unroll
  for (int i = 0; i < 4; ++i) {
    int idx = threadIdx.x + 256 * i;      // 1024 elements
    int r = idx >> 5, c = idx & 31;       // r = k offset, c = n offset
    tile[r][c] = (__bf16)W[(size_t)(k0 + r) * N + n0 + c];
  }
  __syncthreads();
#pragma unroll
  for (int i = 0; i < 4; ++i) {
    int idx = threadIdx.x + 256 * i;
    int r = idx >> 5, c = idx & 31;       // r = n offset, c = k offset
    Wt[(size_t)(n0 + r) * K + k0 + c] = tile[c][r];
  }
}

// ---------------------------------------------------------------------------
// GEMM: C[4096 x Nw] = A[4096 x 1024](bf16) * W^T[Nw x 1024](bf16, pre-transposed)
// MODE 0: write bf16 Q/K/V de-interleaved to [B,H,T,E]
// MODE 1: write fp32 out[4096 x 64] + bias
// Block: 256 threads (8 waves), tile 128(M) x 64(N), K-step 32.
// Both tiles are contiguous bf16 -> async DMA straight into LDS.
// ---------------------------------------------------------------------------
template <int MODE>
__global__ __launch_bounds__(256) void gemm_proj(const __bf16* __restrict__ A,
                                                 const __bf16* __restrict__ Wt,
                                                 const float* __restrict__ bias,
                                                 __bf16* __restrict__ outB,
                                                 float* __restrict__ outF) {
  __shared__ __align__(16) __bf16 As[128][32];   // A tile (row-major)
  __shared__ __align__(16) __bf16 Bst[64][32];   // W^T tile: [n][k]

  const int tid   = threadIdx.x;
  const int wave  = tid >> 5;
  const int lane  = tid & 31;
  const int lm    = lane & 15;
  const int hi    = lane >> 4;
  const int mBase = blockIdx.x * 128;
  const int nBase = blockIdx.y * 64;

  v8f acc[4] = {v8f{}, v8f{}, v8f{}, v8f{}};

  for (int k0 = 0; k0 < DIN; k0 += 32) {
    __syncthreads();
#ifdef HAVE_ASYNC_LDS
    // A tile: 128x32 bf16 = 512 16B chunks (2 issues/thread)
#pragma unroll
    for (int i = 0; i < 2; ++i) {
      int idx = tid + 256 * i;
      int r = idx >> 2, c8 = (idx & 3) * 8;
      const __bf16* g = A + (size_t)(mBase + r) * DIN + k0 + c8;
      __builtin_amdgcn_global_load_async_to_lds_b128(GPTR(g), LPTR(&As[r][c8]), 0, 0);
    }
    // W^T tile: 64x32 bf16 = 256 16B chunks (1 issue/thread)
    {
      int n = tid >> 2, c8 = (tid & 3) * 8;
      const __bf16* g = Wt + (size_t)(nBase + n) * DIN + k0 + c8;
      __builtin_amdgcn_global_load_async_to_lds_b128(GPTR(g), LPTR(&Bst[n][c8]), 0, 0);
    }
#else
#pragma unroll
    for (int i = 0; i < 2; ++i) {
      int idx = tid + 256 * i;
      int r = idx >> 2, c8 = (idx & 3) * 8;
      *(v8bf*)&As[r][c8] = *(const v8bf*)(A + (size_t)(mBase + r) * DIN + k0 + c8);
    }
    {
      int n = tid >> 2, c8 = (tid & 3) * 8;
      *(v8bf*)&Bst[n][c8] = *(const v8bf*)(Wt + (size_t)(nBase + n) * DIN + k0 + c8);
    }
#endif
    // Prefetch next A tile slab (global_prefetch_b8 path on gfx1250)
    if (k0 + 32 < DIN)
      __builtin_prefetch(A + (size_t)(mBase + (tid >> 1)) * DIN + k0 + 32, 0, 1);
#ifdef HAVE_ASYNC_LDS
    __builtin_amdgcn_s_wait_asynccnt(0);
#endif
    __syncthreads();

    const int m = (wave << 4) + lm;       // wave owns rows [wave*16, wave*16+16)
    v16bf a = make_afrag(&As[m][0], hi);
#pragma unroll
    for (int j = 0; j < 4; ++j) {
      v16bf b = make_bfrag(&Bst[(j << 4) + lm][hi << 4]);
      acc[j] = wmma_bf16(a, b, acc[j]);
    }
  }

  // Writeback. C/D layout: lane l, elem r -> row (l<16 ? r : r+8), col l%16.
#pragma unroll
  for (int j = 0; j < 4; ++j) {
#pragma unroll
    for (int r = 0; r < 8; ++r) {
      int rowLoc = (wave << 4) + ((lane < 16) ? r : r + 8);
      int colLoc = (j << 4) + lm;
      int bt = mBase + rowLoc;
      if constexpr (MODE == 0) {
        int c = nBase + colLoc;
        int b = bt >> 11, t = bt & (T_ - 1);
        int h = c & (H_ - 1), e = c >> 4;   // interleaved reshape (E,H), H fastest
        outB[(((size_t)(b * H_ + h)) * T_ + t) * E_ + e] = (__bf16)acc[j][r];
      } else {
        outF[(size_t)bt * E_ + colLoc] = acc[j][r] + bias[colLoc];
      }
    }
  }
}

// ---------------------------------------------------------------------------
// Flash attention. One block = (b, h, 64 query rows); 4 waves x 16 rows each.
// Key tiles of 32; S = Q K^T (2 wmma per 16-key group), online softmax,
// O += P V (4 wmma). Causal mask: score -= 1e9 when key > query (matches ref).
// K and V tiles are async-DMA'd once per block into LDS in natural [key][e]
// layout; PV B-fragments gather V columns with strided LDS reads.
// ---------------------------------------------------------------------------
__global__ __launch_bounds__(128) void attn_flash(const __bf16* __restrict__ Qb,
                                                  const __bf16* __restrict__ Kb,
                                                  const __bf16* __restrict__ Vb,
                                                  __bf16* __restrict__ Ob) {
  __shared__ __align__(16) __bf16 Ksh[32][64];      // K tile, natural [key][e]
  __shared__ __align__(16) __bf16 Vsh[32][64];      // V tile, natural [key][e]
  __shared__ __align__(16) __bf16 Psh[4][16][32];   // per-wave P tile relayout

  const int tid  = threadIdx.x;
  const int wave = tid >> 5;
  const int lane = tid & 31;
  const int lm   = lane & 15;
  const int hi   = lane >> 4;
  const int qt   = blockIdx.x * 64;
  const int h    = blockIdx.y;
  const int b    = blockIdx.z;

  const size_t headOff = ((size_t)(b * H_ + h)) * T_ * E_;
  const __bf16* Qh = Qb + headOff;
  const __bf16* Kh = Kb + headOff;
  const __bf16* Vh = Vb + headOff;

  // Q fragments for this wave's 16 rows, k-dim chunks [0,32) and [32,64)
  const __bf16* qrow = Qh + (size_t)(qt + (wave << 4) + lm) * E_;
  v16bf aq0 = make_afrag(qrow, hi);
  v16bf aq1 = make_afrag(qrow + 32, hi);

  float mrow[8], lrow[8];
#pragma unroll
  for (int r = 0; r < 8; ++r) { mrow[r] = -1e30f; lrow[r] = 0.0f; }
  v8f acc[4] = {v8f{}, v8f{}, v8f{}, v8f{}};

  const int nTiles = (qt + 64) >> 5;   // causal: keys up to end of query tile
  for (int kt = 0; kt < nTiles; ++kt) {
    const int s0 = kt << 5;
    __syncthreads();
    // Stage K and V tiles (32 keys x 64 e each), layout preserved
#ifdef HAVE_ASYNC_LDS
#pragma unroll
    for (int i = 0; i < 2; ++i) {
      int idx = tid + 128 * i;          // 256 chunks of 8 bf16 each
      int k = idx >> 3, e8 = (idx & 7) * 8;
      const size_t off = (size_t)(s0 + k) * E_ + e8;
      __builtin_amdgcn_global_load_async_to_lds_b128(GPTR(Kh + off),
                                                     LPTR(&Ksh[k][e8]), 0, 0);
      __builtin_amdgcn_global_load_async_to_lds_b128(GPTR(Vh + off),
                                                     LPTR(&Vsh[k][e8]), 0, 0);
    }
    __builtin_amdgcn_s_wait_asynccnt(0);
#else
#pragma unroll
    for (int i = 0; i < 2; ++i) {
      int idx = tid + 128 * i;
      int k = idx >> 3, e8 = (idx & 7) * 8;
      const size_t off = (size_t)(s0 + k) * E_ + e8;
      *(v8bf*)&Ksh[k][e8] = *(const v8bf*)(Kh + off);
      *(v8bf*)&Vsh[k][e8] = *(const v8bf*)(Vh + off);
    }
#endif
    __syncthreads();

    // Scores for two 16-key groups; K^T fragments read contiguous e from LDS
    v8f s[2];
#pragma unroll
    for (int g = 0; g < 2; ++g) {
      const __bf16* krow = &Ksh[(g << 4) + lm][0];
      v16bf bk0 = make_bfrag(krow + (hi << 4));        // k-dim 0..31
      v16bf bk1 = make_bfrag(krow + 32 + (hi << 4));   // k-dim 32..63
      v8f sc = v8f{};
      sc = wmma_bf16(aq0, bk0, sc);
      sc = wmma_bf16(aq1, bk1, sc);
#pragma unroll
      for (int r = 0; r < 8; ++r) {
        int row = qt + (wave << 4) + ((lane < 16) ? r : r + 8);
        int key = s0 + (g << 4) + lm;
        float v = sc[r] * 0.125f;                      // 1/sqrt(64)
        if (key > row) v -= 1e9f;
        sc[r] = v;
      }
      s[g] = sc;
    }

    // Online softmax: row stats live replicated across the 16 lanes of a half
#pragma unroll
    for (int r = 0; r < 8; ++r) {
      float tm = fmaxf(s[0][r], s[1][r]);
#pragma unroll
      for (int off = 8; off; off >>= 1) tm = fmaxf(tm, __shfl_xor(tm, off, 16));
      float mn = fmaxf(mrow[r], tm);
      float corr = __expf(mrow[r] - mn);
      mrow[r] = mn;
      float p0 = __expf(s[0][r] - mn);
      float p1 = __expf(s[1][r] - mn);
      s[0][r] = p0; s[1][r] = p1;
      float ps = p0 + p1;
#pragma unroll
      for (int off = 8; off; off >>= 1) ps += __shfl_xor(ps, off, 16);
      lrow[r] = lrow[r] * corr + ps;
#pragma unroll
      for (int j = 0; j < 4; ++j) acc[j][r] *= corr;
    }

    // Relayout P: C-layout regs -> LDS -> A-layout fragment (intra-wave only)
#pragma unroll
    for (int g = 0; g < 2; ++g)
#pragma unroll
      for (int r = 0; r < 8; ++r)
        Psh[wave][(lane < 16) ? r : r + 8][(g << 4) + lm] = (__bf16)s[g][r];
    v16bf pf = make_afrag(&Psh[wave][lm][0], hi);

    // O += P * V. B-frag elem i = V[s0 + hi*16 + i][e]: strided gather from LDS.
#pragma unroll
    for (int j = 0; j < 4; ++j) {
      v16bf bv;
#pragma unroll
      for (int i = 0; i < 16; ++i) bv[i] = Vsh[(hi << 4) + i][(j << 4) + lm];
      acc[j] = wmma_bf16(pf, bv, acc[j]);
    }
  }

  // Normalize and write O interleaved: flat col = e*H + h  (matches reference)
#pragma unroll
  for (int j = 0; j < 4; ++j) {
#pragma unroll
    for (int r = 0; r < 8; ++r) {
      int row = qt + (wave << 4) + ((lane < 16) ? r : r + 8);
      int e = (j << 4) + lm;
      float v = acc[j][r] / lrow[r];
      Ob[((size_t)(b * T_) + row) * EH + (size_t)e * H_ + h] = (__bf16)v;
    }
  }
}

// ---------------------------------------------------------------------------
extern "C" void kernel_launch(void* const* d_in, const int* in_sizes, int n_in,
                              void* d_out, int out_size, void* d_ws, size_t ws_size,
                              hipStream_t stream) {
  const float* inp = (const float*)d_in[0];
  // d_in[1] = mask (unused: causal mask computed analytically)
  const float* Wq = (const float*)d_in[2];
  const float* Wk = (const float*)d_in[3];
  const float* Wv = (const float*)d_in[4];
  const float* Wu = (const float*)d_in[5];
  const float* bu = (const float*)d_in[6];
  float* out = (float*)d_out;

  const size_t MTOK   = (size_t)B_ * T_;                // 4096 tokens
  const size_t QKV_E  = MTOK * EH / H_ * H_;            // 4,194,304 elems
  __bf16* Abf  = (__bf16*)d_ws;                         // [4096,1024]
  __bf16* Wqt  = Abf  + MTOK * DIN;                     // [1024,1024] W^T
  __bf16* Wkt  = Wqt  + (size_t)EH * DIN;
  __bf16* Wvt  = Wkt  + (size_t)EH * DIN;
  __bf16* Wut  = Wvt  + (size_t)EH * DIN;               // [64,1024] W^T
  __bf16* Qb   = Wut  + (size_t)E_ * DIN;               // [B,H,T,E]
  __bf16* Kb   = Qb   + QKV_E;
  __bf16* Vb   = Kb   + QKV_E;
  __bf16* Ob   = Vb   + QKV_E;                          // [B,T,1024] interleaved

  dim3 blk(256);

  // Stage 0: one-shot conversions (inp -> bf16; weights -> bf16 transposed)
  convert_bf16<<<dim3((MTOK * DIN) / 2048), blk, 0, stream>>>(inp, Abf);
  transpose_convert<<<dim3(DIN / 32, EH / 32), blk, 0, stream>>>(Wq, Wqt, DIN, EH);
  transpose_convert<<<dim3(DIN / 32, EH / 32), blk, 0, stream>>>(Wk, Wkt, DIN, EH);
  transpose_convert<<<dim3(DIN / 32, EH / 32), blk, 0, stream>>>(Wv, Wvt, DIN, EH);
  transpose_convert<<<dim3(DIN / 32, E_ / 32), blk, 0, stream>>>(Wu, Wut, DIN, E_);

  // Stage 1: Q/K/V projections
  dim3 gProj(32, 16, 1);   // M=4096/128, N=1024/64
  gemm_proj<0><<<gProj, blk, 0, stream>>>(Abf, Wqt, nullptr, Qb, nullptr);
  gemm_proj<0><<<gProj, blk, 0, stream>>>(Abf, Wkt, nullptr, Kb, nullptr);
  gemm_proj<0><<<gProj, blk, 0, stream>>>(Abf, Wvt, nullptr, Vb, nullptr);

  // Stage 2: flash attention
  attn_flash<<<dim3(T_ / 64, H_, B_), dim3(128), 0, stream>>>(Qb, Kb, Vb, Ob);

  // Stage 3: output projection + bias
  gemm_proj<1><<<dim3(32, 1, 1), blk, 0, stream>>>(Ob, Wut, bu, nullptr, out);
}
